// QuarotFP16Linear_63213328662919
// MI455X (gfx1250) — matrix operand order repaired and checked by reading, not compiled, not run on hardware
//
#include <hip/hip_runtime.h>

typedef __attribute__((ext_vector_type(16))) __bf16 v16bf;
typedef __attribute__((ext_vector_type(8)))  __bf16 v8bf;
typedef __attribute__((ext_vector_type(4)))  __bf16 v4bf;
typedef __attribute__((ext_vector_type(8)))  float  v8f;
typedef __attribute__((ext_vector_type(4)))  float  v4f;

namespace {
constexpr int BM = 128, BN = 128, BK = 64;
constexpr int LDK = 72;               // padded LDS row stride in bf16 elems (144 B)
constexpr int M_DIM = 4 * 2048;       // B*S tokens = 8192
constexpr int N_DIM = 11008;          // OUT
constexpr int K_DIM = 4096;           // IN
constexpr int GQ    = 32;             // quant groups per row (K / 128)
constexpr int NK    = K_DIM / BK;     // 64 k-stages
constexpr int GROUP_M = 16;           // L2-reuse block swizzle
}

union Frag { v16bf v; v8bf h[2]; };

__global__ __launch_bounds__(256)
void qlinear_bf16_wmma(const float* __restrict__ qx,
                       const float* __restrict__ sx,
                       const float* __restrict__ wq,
                       const float* __restrict__ wsc,
                       const float* __restrict__ wof,
                       float* __restrict__ out)
{
  // double-buffered bf16 operand tiles: 2 * (18 + 18) KB = 72 KB of 320 KB/WGP
  __shared__ __bf16 ldsA[2][BM * LDK];
  __shared__ __bf16 ldsB[2][BN * LDK];

  // ---- block swizzle: sweep N within a 16-wide M group for L2 reuse of A ----
  const int num_n = N_DIM / BN;                        // 86
  int pid   = blockIdx.x;
  int group = pid / (GROUP_M * num_n);
  int rem   = pid - group * (GROUP_M * num_n);
  int pid_m = group * GROUP_M + (rem % GROUP_M);
  int pid_n = rem / GROUP_M;

  const int tid  = threadIdx.x;
  const int lane = tid & 31;
  const int wave = tid >> 5;
  const int lr   = lane & 15;
  const int lh   = lane >> 4;
  const int wm   = wave >> 1;                          // 0..3 : M direction
  const int wn   = wave & 1;                           // 0..1 : N direction

  // global-load mapping: 16x16 thread grid, each thread: 8 rows x one float4
  const int lcol = tid & 15;                           // k offset / 4
  const int lrow = tid >> 4;                           // 0..15

  // bumped base pointers; per-row strides become 24-bit immediate offsets
  const float* aP = qx + (size_t)(pid_m * BM + lrow) * K_DIM + lcol * 4;
  const float* bP = wq + (size_t)(pid_n * BN + lrow) * K_DIM + lcol * 4;
  const float* wscP = wsc + (pid_n * BN + lrow) * GQ;
  const float* wofP = wof + (pid_n * BN + lrow) * GQ;

  v4f   areg[8], breg[8];
  float asc[8], bsc[8], bof[8];

  // k-invariant activation scales: load once
  #pragma unroll
  for (int i = 0; i < 8; ++i) asc[i] = sx[pid_m * BM + lrow + 16 * i];

  v8f acc[2][4] = {};

  auto load_quant = [&](int g) {       // weight group scales: every other stage
    #pragma unroll
    for (int i = 0; i < 8; ++i) {
      bsc[i] = wscP[16 * i * GQ + g];
      bof[i] = wofP[16 * i * GQ + g];
    }
  };

  auto load_stage = [&]() {            // global fp32 tiles -> regs, bump ptrs
    #pragma unroll
    for (int i = 0; i < 8; ++i) {
      areg[i] = *(const v4f*)(aP + (size_t)16 * i * K_DIM);
      breg[i] = *(const v4f*)(bP + (size_t)16 * i * K_DIM);
    }
    aP += BK;
    bP += BK;
  };

  auto store_stage = [&](int buf) {    // dequant + f32->bf16 + LDS store
    #pragma unroll
    for (int i = 0; i < 8; ++i) {
      const int r = lrow + 16 * i;
      v4f av = areg[i] * asc[i];                       // x * scale_x
      v4f bv = (breg[i] - bof[i]) * bsc[i];            // (q - off) * wscale
      v4bf a4 = { (__bf16)av.x, (__bf16)av.y, (__bf16)av.z, (__bf16)av.w };
      v4bf b4 = { (__bf16)bv.x, (__bf16)bv.y, (__bf16)bv.z, (__bf16)bv.w };
      *(v4bf*)&ldsA[buf][r * LDK + lcol * 4] = a4;
      *(v4bf*)&ldsB[buf][r * LDK + lcol * 4] = b4;
    }
  };

  auto compute = [&](int buf) {
    #pragma unroll
    for (int ks = 0; ks < 2; ++ks) {   // two 16x16x32 substeps per stage
      Frag a[2]; Frag b[4];
      const int ka = ks * 32 + lh * 16; // A: lanes 0-15 K0..15, lanes 16-31 K16..31
      #pragma unroll
      for (int mt = 0; mt < 2; ++mt) {
        const int row = wm * 32 + mt * 16 + lr;
        a[mt].h[0] = *(const v8bf*)&ldsA[buf][row * LDK + ka];
        a[mt].h[1] = *(const v8bf*)&ldsA[buf][row * LDK + ka + 8];
      }
      const int kb = ks * 32 + lh * 16; // B: lanes 0-15 K0..15, lanes 16-31 K16..31
      #pragma unroll
      for (int nt = 0; nt < 4; ++nt) {
        const int col = wn * 64 + nt * 16 + lr;
        b[nt].h[0] = *(const v8bf*)&ldsB[buf][col * LDK + kb];
        b[nt].h[1] = *(const v8bf*)&ldsB[buf][col * LDK + kb + 8];
      }
      #pragma unroll
      for (int mt = 0; mt < 2; ++mt)
        #pragma unroll
        for (int nt = 0; nt < 4; ++nt)
          acc[mt][nt] = __builtin_amdgcn_wmma_f32_16x16x32_bf16(
              false, a[mt].v, false, b[nt].v, (short)0, acc[mt][nt],
              false, false);
    }
  };

  // ---- pipeline prologue ----
  load_quant(0);
  load_stage();                        // stage 0 -> regs
  store_stage(0);                      // stage 0 -> buf 0
  load_stage();                        // stage 1 -> regs
  __syncthreads();

  // ---- main loop: 1 barrier/stage; stores to buf^1 overlap compute on buf ----
  for (int kt = 0; kt < NK; ++kt) {
    const int buf = kt & 1;
    if (kt + 1 < NK) {
      if (((kt + 1) & 1) == 0) load_quant((kt + 1) >> 1);  // group = stage/2
      store_stage(buf ^ 1);            // stage kt+1 regs -> other buffer
    }
    if (kt + 2 < NK) load_stage();     // stage kt+2 global loads overlap compute
    compute(buf);
    __syncthreads();
  }

  // ---- epilogue: C layout VGPR i -> M = lh*8 + i, N = lr ----
  #pragma unroll
  for (int mt = 0; mt < 2; ++mt) {
    #pragma unroll
    for (int nt = 0; nt < 4; ++nt) {
      const int row0 = pid_m * BM + wm * 32 + mt * 16 + lh * 8;
      const int col  = pid_n * BN + wn * 64 + nt * 16 + lr;
      #pragma unroll
      for (int i = 0; i < 8; ++i)
        out[(size_t)(row0 + i) * N_DIM + col] = acc[mt][nt][i];
    }
  }
}

extern "C" void kernel_launch(void* const* d_in, const int* in_sizes, int n_in,
                              void* d_out, int out_size, void* d_ws, size_t ws_size,
                              hipStream_t stream) {
  const float* qx  = (const float*)d_in[0];   // quantized_x  [B,S,IN]  f32
  const float* sx  = (const float*)d_in[1];   // scales_x     [B,S,1]   f32
  const float* wq  = (const float*)d_in[2];   // weight       [OUT,IN]  f32
  const float* wsc = (const float*)d_in[3];   // weight_scales[OUT,G]   f32
  const float* wof = (const float*)d_in[4];   // offset_w     [OUT,G]   f32
  float* out = (float*)d_out;                 // [B,S,OUT] f32

  dim3 grid((M_DIM / BM) * (N_DIM / BN));     // 64 * 86 = 5504 blocks
  qlinear_bf16_wmma<<<grid, 256, 0, stream>>>(qx, sx, wq, wsc, wof, out);
}
